// GestureModel_64536178590131
// MI455X (gfx1250) — matrix operand order repaired and checked by reading, not compiled, not run on hardware
//
#include <hip/hip_runtime.h>

// CT-RNN persistent-scan kernel for MI455X (gfx1250, wave32, WMMA).
// B=512, T=1024, I=64, H=256, C=5, UNFOLDS=6, dt=0.1, tau=1.
//
// 32 blocks x 512 threads (16 waves). Block owns a 16-row batch tile for the
// whole T sequence. Wave n owns H-columns [16n,16n+16); Wh/Wx B-fragments are
// VGPR-resident for the whole kernel; h lives in f32 C/D-layout registers with
// a ping-pong f16 LDS mirror feeding the WMMA A operand (1 barrier/unfold).
// x is double-buffered in LDS and staged for t+1 while t computes.

typedef __attribute__((ext_vector_type(16))) _Float16 v16h;
typedef __attribute__((ext_vector_type(8)))  _Float16 v8h;
typedef __attribute__((ext_vector_type(8)))  float    v8f;

#if __has_builtin(__builtin_amdgcn_tanhf)
#define TANH_F32(x) __builtin_amdgcn_tanhf(x)
#else
#define TANH_F32(x) tanhf(x)
#endif

constexpr int kB = 512, kT = 1024, kI = 64, kH = 256, kC = 5;
constexpr int kUNFOLDS = 6;                // even -> static ping-pong parity
constexpr float kDT = 0.1f;
constexpr int BTILE = 16;
constexpr int NWAVES = kH / 16;            // 16 waves
constexpr int THREADS = NWAVES * 32;       // 512 threads

union V16U { v16h v; v8h h[2]; };

__global__ __launch_bounds__(THREADS, 1)
void ctrnn_scan_kernel(const float* __restrict__ x,
                       const float* __restrict__ W,
                       const float* __restrict__ b,
                       const float* __restrict__ Wfc,
                       const float* __restrict__ bfc,
                       float* __restrict__ out)
{
    __shared__ _Float16 sh_h[2][BTILE][kH];   // 16 KB : ping-pong f16 h tile
    __shared__ _Float16 sh_x[2][BTILE][kI];   //  4 KB : double-buffered x tile
    __shared__ _Float16 sh_wfc[8][32][16];    //  8 KB : W_fc fragments (N padded)

    const int tid   = threadIdx.x;
    const int wave  = tid >> 5;
    const int lane  = tid & 31;
    const int lhalf = lane & 15;
    const int hi    = lane >> 4;              // lane half: 0 / 1
    const int b0    = blockIdx.x * BTILE;
    const int ncol  = wave * 16 + lhalf;      // this wave/lane's H column

    // ---- one-time LDS init ------------------------------------------------
    for (int i = tid; i < 2 * BTILE * kH; i += THREADS)
        ((_Float16*)sh_h)[i] = (_Float16)0.0f;
    for (int i = tid; i < 8 * 32 * 16; i += THREADS) {
        const int e  = i & 15;
        const int ln = (i >> 4) & 31;
        const int kc = i >> 9;
        const int K  = kc * 32 + ((ln >= 16) ? 8 : 0) + (e & 7) + (e >> 3) * 16;
        const int c  = ln & 15;
        sh_wfc[kc][ln][e] = (_Float16)((c < kC) ? Wfc[K * kC + c] : 0.0f);
    }

    // ---- per-lane constants ----------------------------------------------
    const float bias_val = b[ncol];
    const float bfc_val  = (lhalf < kC) ? bfc[lhalf] : 0.0f;

    // Wx B-fragments (K = 0..63, 2 chunks of 32) -> registers, f16
    V16U wxf[2];
#pragma unroll
    for (int kc = 0; kc < 2; ++kc)
#pragma unroll
        for (int e = 0; e < 16; ++e) {
            const int K = kc * 32 + hi * 8 + (e & 7) + (e >> 3) * 16;
            wxf[kc].v[e] = (_Float16)W[K * kH + ncol];
        }

    // Wh B-fragments (W rows 64..319, 8 chunks of 32) -> 64 VGPRs, f16
    V16U whf[8];
#pragma unroll
    for (int kc = 0; kc < 8; ++kc)
#pragma unroll
        for (int e = 0; e < 16; ++e) {
            const int K = kI + kc * 32 + hi * 8 + (e & 7) + (e >> 3) * 16;
            whf[kc].v[e] = (_Float16)W[K * kH + ncol];
        }

    // h slice in f32 accumulator (C/D) layout: element (M = v + 8*hi, N = lhalf)
    float hreg[8];
#pragma unroll
    for (int v = 0; v < 8; ++v) hreg[v] = 0.0f;

    // x staging: 512 threads x 2 floats = 16x64 tile
    const int xrow = tid >> 5;
    const int xcol = (tid & 31) * 2;
    const float* xbase = x + (size_t)(b0 + xrow) * kT * kI + xcol;

    // stage x for t = 0
    {
        const float2 xv = *(const float2*)(xbase);
        sh_x[0][xrow][xcol]     = (_Float16)xv.x;
        sh_x[0][xrow][xcol + 1] = (_Float16)xv.y;
    }
    __syncthreads();

    // holds bias + x_t @ Wx for current t (C/D layout)
    v8f xw;

    // one CT-RNN unfold: read h from src, write new h to dst, one barrier.
    // Two independent WMMA accumulator chains for ILP.
    auto unfold_round = [&](const _Float16 (&src)[BTILE][kH],
                            _Float16 (&dst)[BTILE][kH]) {
        v8f a0 = xw;
        v8f a1;
#pragma unroll
        for (int i = 0; i < 8; ++i) a1[i] = 0.0f;
#pragma unroll
        for (int kc = 0; kc < 8; kc += 2) {
            V16U fa, fb;
            const int k0 = kc * 32 + hi * 8;
            const int k1 = (kc + 1) * 32 + hi * 8;
            fa.h[0] = *(const v8h*)&src[lhalf][k0];
            fa.h[1] = *(const v8h*)&src[lhalf][k0 + 16];
            fb.h[0] = *(const v8h*)&src[lhalf][k1];
            fb.h[1] = *(const v8h*)&src[lhalf][k1 + 16];
            a0 = __builtin_amdgcn_wmma_f32_16x16x32_f16(false, fa.v, false, whf[kc].v,
                                                        (short)0, a0, false, false);
            a1 = __builtin_amdgcn_wmma_f32_16x16x32_f16(false, fb.v, false, whf[kc + 1].v,
                                                        (short)0, a1, false, false);
        }
#pragma unroll
        for (int v = 0; v < 8; ++v)
            hreg[v] = (1.0f - kDT) * hreg[v] + kDT * TANH_F32(a0[v] + a1[v]);
#pragma unroll
        for (int v = 0; v < 8; ++v)
            dst[v + hi * 8][wave * 16 + lhalf] = (_Float16)hreg[v];
        __syncthreads();
    };

    for (int t = 0; t < kT; ++t) {
        const int xp = t & 1;

        // ---- xw = bias + x_tile @ Wx  (2 WMMAs, reads sh_x[xp]) -----------
#pragma unroll
        for (int i = 0; i < 8; ++i) xw[i] = bias_val;
#pragma unroll
        for (int kc = 0; kc < 2; ++kc) {
            V16U a;
            const int koff = kc * 32 + hi * 8;
            a.h[0] = *(const v8h*)&sh_x[xp][lhalf][koff];
            a.h[1] = *(const v8h*)&sh_x[xp][lhalf][koff + 16];
            xw = __builtin_amdgcn_wmma_f32_16x16x32_f16(false, a.v, false, wxf[kc].v,
                                                        (short)0, xw, false, false);
        }

        // ---- stage x for t+1 into the other buffer (no barrier needed:
        //      visibility is covered by the unfold barriers below) ----------
        if (t + 1 < kT) {
            const float2 xv = *(const float2*)(xbase + (size_t)(t + 1) * kI);
            sh_x[xp ^ 1][xrow][xcol]     = (_Float16)xv.x;
            sh_x[xp ^ 1][xrow][xcol + 1] = (_Float16)xv.y;
            if (t + 2 < kT)
                __builtin_prefetch(xbase + (size_t)(t + 2) * kI, 0, 1); // global_prefetch_b8
        }

        // ---- 6 unfolds, ping-pong h buffers, 1 barrier each ---------------
#pragma unroll 1
        for (int up = 0; up < kUNFOLDS / 2; ++up) {
            unfold_round(sh_h[0], sh_h[1]);
            unfold_round(sh_h[1], sh_h[0]);
        }
        // h now lives in sh_h[0] (and in hreg)

        // ---- per-step FC (16x256 @ 256x5 padded to 16) on wave 0 ----------
        if (wave == 0) {
            v8f f0, f1;
#pragma unroll
            for (int i = 0; i < 8; ++i) { f0[i] = 0.0f; f1[i] = 0.0f; }
#pragma unroll
            for (int kc = 0; kc < 8; kc += 2) {
                V16U a0, a1, bw0, bw1;
                const int k0 = kc * 32 + hi * 8;
                const int k1 = (kc + 1) * 32 + hi * 8;
                a0.h[0]  = *(const v8h*)&sh_h[0][lhalf][k0];
                a0.h[1]  = *(const v8h*)&sh_h[0][lhalf][k0 + 16];
                a1.h[0]  = *(const v8h*)&sh_h[0][lhalf][k1];
                a1.h[1]  = *(const v8h*)&sh_h[0][lhalf][k1 + 16];
                bw0.h[0] = *(const v8h*)&sh_wfc[kc][lane][0];
                bw0.h[1] = *(const v8h*)&sh_wfc[kc][lane][8];
                bw1.h[0] = *(const v8h*)&sh_wfc[kc + 1][lane][0];
                bw1.h[1] = *(const v8h*)&sh_wfc[kc + 1][lane][8];
                f0 = __builtin_amdgcn_wmma_f32_16x16x32_f16(false, a0.v, false, bw0.v,
                                                            (short)0, f0, false, false);
                f1 = __builtin_amdgcn_wmma_f32_16x16x32_f16(false, a1.v, false, bw1.v,
                                                            (short)0, f1, false, false);
            }
            if (lhalf < kC) {
#pragma unroll
                for (int v = 0; v < 8; ++v) {
                    const int m = v + hi * 8;
                    out[(size_t)(b0 + m) * kC * kT + (size_t)lhalf * kT + t] =
                        f0[v] + f1[v] + bfc_val;
                }
            }
        }
    }
}

extern "C" void kernel_launch(void* const* d_in, const int* in_sizes, int n_in,
                              void* d_out, int out_size, void* d_ws, size_t ws_size,
                              hipStream_t stream) {
    (void)in_sizes; (void)n_in; (void)out_size; (void)d_ws; (void)ws_size;
    const float* x   = (const float*)d_in[0];
    const float* W   = (const float*)d_in[1];
    const float* b   = (const float*)d_in[2];
    const float* Wfc = (const float*)d_in[3];
    const float* bfc = (const float*)d_in[4];
    float* out = (float*)d_out;

    dim3 grid(kB / BTILE);   // 32 workgroups, one 16-row batch tile each
    ctrnn_scan_kernel<<<grid, THREADS, 0, stream>>>(x, W, b, Wfc, bfc, out);
}
